// Qwen2MoTDecoderLayer_18923625906681
// MI455X (gfx1250) — compile-verified, matrix operand order checked by reading, not run on hardware
//
#include <hip/hip_runtime.h>
#include <hip/hip_bf16.h>

typedef __bf16 v16bf __attribute__((ext_vector_type(16)));
typedef __bf16 v8bf  __attribute__((ext_vector_type(8)));
typedef __bf16 v4bf  __attribute__((ext_vector_type(4)));
typedef float  v8f   __attribute__((ext_vector_type(8)));

// ---------------------------------------------------------------------------
// Problem constants (from the reference)
// ---------------------------------------------------------------------------
constexpr int Bc   = 16;
constexpr int Lc   = 512;
constexpr int Sc   = Bc * Lc;      // 8192
constexpr int Hc   = 16;
constexpr int HKVc = 4;
constexpr int Dc   = 128;
constexpr int HIDc = 2048;
constexpr int HD   = Hc * Dc;      // 2048
constexpr int KVD  = HKVc * Dc;    // 512
constexpr int MH   = Sc / 2;       // 4096 tokens per branch

// ---------------------------------------------------------------------------
// WMMA helper
// ---------------------------------------------------------------------------
__device__ __forceinline__ v8f wmma_bf16(v16bf a, v16bf b, v8f c) {
  return __builtin_amdgcn_wmma_f32_16x16x32_bf16(
      false, a, false, b, (short)0, c, false, false);
}

// Load a 16x32 bf16 fragment (A layout, and B layout for NT operands) from a
// row-major bf16 source with leading dimension `ld` (elements). Per the CDNA5
// layout: lanes 0-15 hold K{0..7,16..23}, lanes 16-31 hold K{8..15,24..31};
// lane%16 selects the row.
__device__ __forceinline__ v16bf load_frag_bf16(const __bf16* __restrict__ base,
                                                int ld, int lane) {
  const int half = lane >> 4;
  const int r    = lane & 15;
  const __bf16* p = base + (size_t)r * (size_t)ld + half * 8;
  v8bf lo = *(const v8bf*)(p);
  v8bf hi = *(const v8bf*)(p + 16);
  v16bf f;
#pragma unroll
  for (int e = 0; e < 8; ++e) { f[e] = lo[e]; f[8 + e] = hi[e]; }
  return f;
}

// ---------------------------------------------------------------------------
// f32 -> bf16 conversion (n must be a multiple of 4; all our sizes are)
// ---------------------------------------------------------------------------
__global__ void __launch_bounds__(256)
cvt_f32_bf16(const float* __restrict__ x, __bf16* __restrict__ y, int n) {
  const int i = (blockIdx.x * 256 + threadIdx.x) * 4;
  if (i < n) {
    float4 v = *(const float4*)(x + i);
    v4bf o;
    o[0] = (__bf16)v.x; o[1] = (__bf16)v.y;
    o[2] = (__bf16)v.z; o[3] = (__bf16)v.w;
    *(v4bf*)(y + i) = o;
  }
}

// ---------------------------------------------------------------------------
// Generic NT GEMM: C(f32) = A(bf16, MxK) * W(bf16, NxK)^T (+bias)
// A row i read at A[(a_scale*i + a_off)*K]; C row i at C[(c_scale*i+c_off)*N].
// Block = 8 waves (2 along M x 4 along N). Each wave: 64x32 tile = 4x2 WMMA
// fragments, 8 WMMA per 32-wide K step. The wave's 32x32 W tile is staged
// into LDS with double-buffered global_load_async_to_lds_b128 (CDNA5 async
// path, ASYNCcnt-tracked); the last K-step is peeled so the steady-state
// loop is branch-free. A fragments load straight from global.
// ---------------------------------------------------------------------------
__global__ void __launch_bounds__(256)
gemm_bias_wmma(const __bf16* __restrict__ A, const __bf16* __restrict__ W,
               const float* __restrict__ bias, float* __restrict__ C,
               int K, int N, int a_scale, int a_off, int c_scale, int c_off) {
  __shared__ __align__(16) __bf16 wlds[8][2][32 * 32];  // 32 KB

  const int wave = threadIdx.x >> 5;
  const int lane = threadIdx.x & 31;
  const int half = lane >> 4;
  const int hl   = lane & 15;
  const int wm   = wave & 1;   // 2 waves along M (64 rows each)
  const int wn   = wave >> 1;  // 4 waves along N (32 cols each)
  const int m0   = blockIdx.y * 128 + wm * 64;
  const int n0   = blockIdx.x * 128 + wn * 32;

  const int lda = a_scale * K;
  const __bf16* Abase = A + ((size_t)a_scale * m0 + a_off) * (size_t)K;
  const __bf16* Wrow  = W + (size_t)n0 * (size_t)K;  // 32 rows x K

  const uint32_t wb[2] = {
      (uint32_t)(uintptr_t)&wlds[wave][0][0],   // generic->LDS: addr[31:0]
      (uint32_t)(uintptr_t)&wlds[wave][1][0]};

  // Issue one 32x32 bf16 tile (2 KB) as 4 async b128 ops (512 B each).
  auto prefetch = [&](const __bf16* src, uint32_t ldsbase) {
#pragma unroll
    for (int i = 0; i < 4; ++i) {
      const int fb = i * 512 + lane * 16;  // byte index within tile
      const int r  = fb >> 6;              // row (64 B per row)
      const int cb = fb & 63;              // byte within row
      const char* g = (const char*)src + (size_t)r * (size_t)K * 2 + cb;
      const uint32_t ld = ldsbase + (uint32_t)fb;
      asm volatile("global_load_async_to_lds_b128 %0, %1, off"
                   :: "v"(ld), "v"(g) : "memory");
    }
  };

  v8f acc[4][2] = {};

  // One 32-wide K step: read W frags from LDS buffer `c`, A frags from global.
  auto kstep = [&](int k0, int c) {
    const __bf16* wt = &wlds[wave][c][0];
    v16bf b0 = load_frag_bf16(wt, 32, lane);            // W rows n0..n0+15
    v16bf b1 = load_frag_bf16(wt + 16 * 32, 32, lane);  // W rows n0+16..n0+31

    v16bf a0 = load_frag_bf16(Abase + k0, lda, lane);
    v16bf a1 = load_frag_bf16(Abase + (size_t)16 * lda + k0, lda, lane);
    v16bf a2 = load_frag_bf16(Abase + (size_t)32 * lda + k0, lda, lane);
    v16bf a3 = load_frag_bf16(Abase + (size_t)48 * lda + k0, lda, lane);

    acc[0][0] = wmma_bf16(a0, b0, acc[0][0]);
    acc[0][1] = wmma_bf16(a0, b1, acc[0][1]);
    acc[1][0] = wmma_bf16(a1, b0, acc[1][0]);
    acc[1][1] = wmma_bf16(a1, b1, acc[1][1]);
    acc[2][0] = wmma_bf16(a2, b0, acc[2][0]);
    acc[2][1] = wmma_bf16(a2, b1, acc[2][1]);
    acc[3][0] = wmma_bf16(a3, b0, acc[3][0]);
    acc[3][1] = wmma_bf16(a3, b1, acc[3][1]);
  };

  prefetch(Wrow, wb[0]);
  int cur = 0;
  int k0 = 0;
  // Steady state: branch-free prefetch-next / wait-oldest / compute.
  for (; k0 < K - 32; k0 += 32) {
    asm volatile("s_wait_dscnt 0x0" ::: "memory");  // done reading next buffer
    prefetch(Wrow + k0 + 32, wb[cur ^ 1]);
    asm volatile("s_wait_asynccnt 0x4" ::: "memory");  // oldest tile landed
    kstep(k0, cur);
    cur ^= 1;
  }
  // Peeled final step.
  asm volatile("s_wait_asynccnt 0x0" ::: "memory");
  kstep(k0, cur);

  // C/D layout: VGPR j, lanes 0-15 -> row j (col=lane); lanes 16-31 -> row j+8
#pragma unroll
  for (int mi = 0; mi < 4; ++mi) {
#pragma unroll
    for (int ni = 0; ni < 2; ++ni) {
      const int ncol = n0 + ni * 16 + hl;
      const float badd = bias ? bias[ncol] : 0.0f;
#pragma unroll
      for (int j = 0; j < 8; ++j) {
        const int mrow = m0 + mi * 16 + half * 8 + j;
        C[((size_t)c_scale * mrow + c_off) * (size_t)N + ncol] =
            acc[mi][ni][j] + badd;
      }
    }
  }
}

// ---------------------------------------------------------------------------
// Per-(token, head) RMSNorm over D=128 + RoPE; reads f32, writes bf16 with a
// fold-in output scale (1/sqrt(D) for Q, 1.0 for K). Even tokens use und
// weights, odd tokens gen weights. blockDim = 128.
// ---------------------------------------------------------------------------
__global__ void __launch_bounds__(128)
norm_rope(const float* __restrict__ x, __bf16* __restrict__ y, int nheads,
          const float* __restrict__ w_und, const float* __restrict__ w_gen,
          const float* __restrict__ cosb, const float* __restrict__ sinb,
          float outscale) {
  const int t = blockIdx.x;
  const int h = blockIdx.y;
  const int d = threadIdx.x;
  const size_t idx = ((size_t)t * nheads + h) * Dc + d;
  const float val = x[idx];

  float ss = val * val;
  ss += __shfl_xor(ss, 1, 32);
  ss += __shfl_xor(ss, 2, 32);
  ss += __shfl_xor(ss, 4, 32);
  ss += __shfl_xor(ss, 8, 32);
  ss += __shfl_xor(ss, 16, 32);

  __shared__ float red[4];
  __shared__ float ybuf[128];
  if ((threadIdx.x & 31) == 0) red[threadIdx.x >> 5] = ss;
  __syncthreads();
  const float tot = red[0] + red[1] + red[2] + red[3];
  const float inv = rsqrtf(tot * (1.0f / 128.0f) + 1e-6f);
  const float* w = (t & 1) ? w_gen : w_und;
  const float yv = val * inv * w[d];
  ybuf[d] = yv;
  __syncthreads();
  const float rot = (d < 64) ? -ybuf[d + 64] : ybuf[d - 64];
  const size_t ci = (size_t)t * Dc + d;
  y[idx] = (__bf16)((yv * cosb[ci] + rot * sinb[ci]) * outscale);
}

// ---------------------------------------------------------------------------
// Causal GQA flash attention, one wave per (batch, head, 16-query tile).
// All operands bf16 (Q pre-scaled by 1/sqrt(D)); fp32 accumulation.
// Per 32-key step: 8 WMMA QK^T, shfl_xor online softmax, P transposed via
// per-wave LDS, V staged d-major in LDS, 8 WMMA PV. Output bf16.
// ---------------------------------------------------------------------------
__global__ void __launch_bounds__(128)
attn_wmma(const __bf16* __restrict__ q, const __bf16* __restrict__ k,
          const __bf16* __restrict__ v, __bf16* __restrict__ o) {
  __shared__ __align__(16) __bf16 smem[4 * 4608];  // per wave: V 32x128 + P 16x32
  const int wave = threadIdx.x >> 5;
  const int lane = threadIdx.x & 31;
  const int half = lane >> 4;
  const int hl   = lane & 15;
  __bf16* vtile = smem + wave * 4608;   // [d*32 + t], d-major
  __bf16* ptile = vtile + 4096;         // [row*32 + t]

  const int wid = blockIdx.x * 4 + wave;
  const int qt = wid & 31;
  const int h  = (wid >> 5) & 15;   // full head = kv*G + g
  const int b  = wid >> 9;
  const int kv = h >> 2;
  const int q0 = qt * 16;

  const __bf16* qbase = q + ((size_t)(b * Lc + q0)) * HD + h * Dc;
  v16bf qf[4];
#pragma unroll
  for (int c = 0; c < 4; ++c)
    qf[c] = load_frag_bf16(qbase + c * 32, HD, lane);

  v8f acc[8] = {};
  float mrow[8], lrow[8];
#pragma unroll
  for (int j = 0; j < 8; ++j) { mrow[j] = -3.0e38f; lrow[j] = 0.0f; }

  const int kend = q0 + 16;
  for (int kt = 0; kt < kend; kt += 32) {
    // ---- scores for keys [kt, kt+32) ----
    const __bf16* kbase = k + ((size_t)(b * Lc + kt)) * KVD + kv * Dc;
    v8f s0 = {}, s1 = {};
#pragma unroll
    for (int c = 0; c < 4; ++c) {
      v16bf kf0 = load_frag_bf16(kbase + c * 32, KVD, lane);
      v16bf kf1 = load_frag_bf16(kbase + (size_t)16 * KVD + c * 32, KVD, lane);
      s0 = wmma_bf16(qf[c], kf0, s0);
      s1 = wmma_bf16(qf[c], kf1, s1);
    }

    // ---- causal mask + row max across the 16-lane half ----
    const int col0 = kt + hl;
    const int col1 = kt + 16 + hl;
    float mx[8];
#pragma unroll
    for (int j = 0; j < 8; ++j) {
      const int row = q0 + half * 8 + j;
      if (col0 > row) s0[j] = -3.0e38f;
      if (col1 > row) s1[j] = -3.0e38f;
      float m = fmaxf(s0[j], s1[j]);
      m = fmaxf(m, __shfl_xor(m, 1, 32));
      m = fmaxf(m, __shfl_xor(m, 2, 32));
      m = fmaxf(m, __shfl_xor(m, 4, 32));
      m = fmaxf(m, __shfl_xor(m, 8, 32));
      mx[j] = m;
    }

    // ---- online softmax update ----
#pragma unroll
    for (int j = 0; j < 8; ++j) {
      const float mn = fmaxf(mrow[j], mx[j]);
      const float alpha = __expf(mrow[j] - mn);
      mrow[j] = mn;
      const float p0 = __expf(s0[j] - mn);
      const float p1 = __expf(s1[j] - mn);
      s0[j] = p0; s1[j] = p1;
      float rs = p0 + p1;
      rs += __shfl_xor(rs, 1, 32);
      rs += __shfl_xor(rs, 2, 32);
      rs += __shfl_xor(rs, 4, 32);
      rs += __shfl_xor(rs, 8, 32);
      lrow[j] = lrow[j] * alpha + rs;
#pragma unroll
      for (int c = 0; c < 8; ++c) acc[c][j] *= alpha;
    }

    // ---- stage P (bf16, 16x32 row-major) ----
#pragma unroll
    for (int j = 0; j < 8; ++j) {
      const int row = half * 8 + j;
      ptile[row * 32 + hl]      = (__bf16)s0[j];
      ptile[row * 32 + 16 + hl] = (__bf16)s1[j];
    }

    // ---- stage V tile (32 keys x 128 d) d-major ----
    const __bf16* vb = v + ((size_t)(b * Lc + kt)) * KVD + kv * Dc;
#pragma unroll 4
    for (int t = 0; t < 32; ++t) {
      v4bf x = *(const v4bf*)(vb + (size_t)t * KVD + lane * 4);
      const int d0 = lane * 4;
      vtile[(d0 + 0) * 32 + t] = x[0];
      vtile[(d0 + 1) * 32 + t] = x[1];
      vtile[(d0 + 2) * 32 + t] = x[2];
      vtile[(d0 + 3) * 32 + t] = x[3];
    }
    __builtin_amdgcn_wave_barrier();  // same-wave DS is in-order; block reordering

    // ---- P fragment (A layout) ----
    v16bf pf;
    {
      const __bf16* pp = ptile + hl * 32 + half * 8;
#pragma unroll
      for (int e = 0; e < 8; ++e) { pf[e] = pp[e]; pf[8 + e] = pp[16 + e]; }
    }

    // ---- PV over 8 d-chunks ----
#pragma unroll
    for (int c = 0; c < 8; ++c) {
      const __bf16* vp = vtile + (c * 16 + hl) * 32 + half * 8;
      v16bf vf;
#pragma unroll
      for (int e = 0; e < 8; ++e) { vf[e] = vp[e]; vf[8 + e] = vp[16 + e]; }
      acc[c] = wmma_bf16(pf, vf, acc[c]);
    }
    __builtin_amdgcn_wave_barrier();
  }

  // ---- normalize and write O (bf16, feeds the O-projection GEMM) ----
  __bf16* ob = o + ((size_t)(b * Lc + q0)) * HD + h * Dc;
#pragma unroll
  for (int c = 0; c < 8; ++c) {
    const int d = c * 16 + hl;
#pragma unroll
    for (int j = 0; j < 8; ++j)
      ob[(size_t)(half * 8 + j) * HD + d] = (__bf16)(acc[c][j] / lrow[j]);
  }
}

// ---------------------------------------------------------------------------
// Launch
// ---------------------------------------------------------------------------
extern "C" void kernel_launch(void* const* d_in, const int* in_sizes, int n_in,
                              void* d_out, int out_size, void* d_ws, size_t ws_size,
                              hipStream_t stream) {
  (void)in_sizes; (void)n_in; (void)out_size; (void)ws_size;

  const float* x_und = (const float*)d_in[0];
  const float* x_gen = (const float*)d_in[1];
  const float* q_w   = (const float*)d_in[4];
  const float* q_b   = (const float*)d_in[5];
  const float* k_w   = (const float*)d_in[6];
  const float* k_b   = (const float*)d_in[7];
  const float* v_w   = (const float*)d_in[8];
  const float* v_b   = (const float*)d_in[9];
  const float* o_w   = (const float*)d_in[10];
  const float* q_wg  = (const float*)d_in[11];
  const float* q_bg  = (const float*)d_in[12];
  const float* k_wg  = (const float*)d_in[13];
  const float* k_bg  = (const float*)d_in[14];
  const float* v_wg  = (const float*)d_in[15];
  const float* v_bg  = (const float*)d_in[16];
  const float* o_wg  = (const float*)d_in[17];
  const float* qn_w  = (const float*)d_in[18];
  const float* kn_w  = (const float*)d_in[19];
  const float* qn_wg = (const float*)d_in[20];
  const float* kn_wg = (const float*)d_in[21];
  const float* cosb  = (const float*)d_in[22];
  const float* sinb  = (const float*)d_in[23];

  // ---- workspace carve-up ----
  char* wp = (char*)d_ws;
  auto alloc_f = [&](size_t n) { float* r = (float*)wp; wp += n * sizeof(float); return r; };
  auto alloc_b = [&](size_t n) { __bf16* r = (__bf16*)wp; wp += n * sizeof(__bf16); return r; };

  float*  qbuf  = alloc_f((size_t)Sc * HD);    // f32 QKV outputs (pre-norm)
  float*  kbuf  = alloc_f((size_t)Sc * KVD);
  float*  vbuf  = alloc_f((size_t)Sc * KVD);
  __bf16* xb_u  = alloc_b((size_t)MH * HIDc);  // bf16 activations
  __bf16* xb_g  = alloc_b((size_t)MH * HIDc);
  __bf16* wq_u  = alloc_b((size_t)HD * HIDc);  // bf16 weights (und)
  __bf16* wk_u  = alloc_b((size_t)KVD * HIDc);
  __bf16* wv_u  = alloc_b((size_t)KVD * HIDc);
  __bf16* wo_u  = alloc_b((size_t)HIDc * HD);
  __bf16* wq_g  = alloc_b((size_t)HD * HIDc);  // bf16 weights (gen)
  __bf16* wk_g  = alloc_b((size_t)KVD * HIDc);
  __bf16* wv_g  = alloc_b((size_t)KVD * HIDc);
  __bf16* wo_g  = alloc_b((size_t)HIDc * HD);
  __bf16* qb16  = alloc_b((size_t)Sc * HD);    // bf16 normed+roped Q (scaled)
  __bf16* kb16  = alloc_b((size_t)Sc * KVD);
  __bf16* vb16  = alloc_b((size_t)Sc * KVD);
  __bf16* ab16  = alloc_b((size_t)Sc * HD);    // bf16 attention output

  auto cvt = [&](const float* src, __bf16* dst, size_t n) {
    cvt_f32_bf16<<<dim3((unsigned)((n / 4 + 255) / 256)), 256, 0, stream>>>(src, dst, (int)n);
  };

  // ---- one-shot bf16 conversion of GEMM operands ----
  cvt(x_und, xb_u, (size_t)MH * HIDc);
  cvt(x_gen, xb_g, (size_t)MH * HIDc);
  cvt(q_w,  wq_u, (size_t)HD * HIDc);
  cvt(k_w,  wk_u, (size_t)KVD * HIDc);
  cvt(v_w,  wv_u, (size_t)KVD * HIDc);
  cvt(o_w,  wo_u, (size_t)HIDc * HD);
  cvt(q_wg, wq_g, (size_t)HD * HIDc);
  cvt(k_wg, wk_g, (size_t)KVD * HIDc);
  cvt(v_wg, wv_g, (size_t)KVD * HIDc);
  cvt(o_wg, wo_g, (size_t)HIDc * HD);

  const dim3 gblk(256);
  // ---- QKV projections (scatter to even/odd rows) ----
  gemm_bias_wmma<<<dim3(HD / 128, MH / 128), gblk, 0, stream>>>(
      xb_u, wq_u, q_b, qbuf, HIDc, HD, 1, 0, 2, 0);
  gemm_bias_wmma<<<dim3(HD / 128, MH / 128), gblk, 0, stream>>>(
      xb_g, wq_g, q_bg, qbuf, HIDc, HD, 1, 0, 2, 1);
  gemm_bias_wmma<<<dim3(KVD / 128, MH / 128), gblk, 0, stream>>>(
      xb_u, wk_u, k_b, kbuf, HIDc, KVD, 1, 0, 2, 0);
  gemm_bias_wmma<<<dim3(KVD / 128, MH / 128), gblk, 0, stream>>>(
      xb_g, wk_g, k_bg, kbuf, HIDc, KVD, 1, 0, 2, 1);
  gemm_bias_wmma<<<dim3(KVD / 128, MH / 128), gblk, 0, stream>>>(
      xb_u, wv_u, v_b, vbuf, HIDc, KVD, 1, 0, 2, 0);
  gemm_bias_wmma<<<dim3(KVD / 128, MH / 128), gblk, 0, stream>>>(
      xb_g, wv_g, v_bg, vbuf, HIDc, KVD, 1, 0, 2, 1);

  // ---- RMSNorm + RoPE -> bf16 (Q gets 1/sqrt(D) folded in); V -> bf16 ----
  const float qscale = 0.0883883476483184405f;  // 1/sqrt(128)
  norm_rope<<<dim3(Sc, Hc), 128, 0, stream>>>(qbuf, qb16, Hc, qn_w, qn_wg, cosb, sinb, qscale);
  norm_rope<<<dim3(Sc, HKVc), 128, 0, stream>>>(kbuf, kb16, HKVc, kn_w, kn_wg, cosb, sinb, 1.0f);
  cvt(vbuf, vb16, (size_t)Sc * KVD);

  // ---- attention: B*H*(L/16) = 8192 waves, 4 waves/block ----
  attn_wmma<<<dim3(Bc * Hc * (Lc / 16) / 4), 128, 0, stream>>>(qb16, kb16, vb16, ab16);

  // ---- output projections (gather even/odd rows of ab16) ----
  float* out_und = (float*)d_out;
  float* out_gen = out_und + (size_t)MH * HIDc;
  gemm_bias_wmma<<<dim3(HIDc / 128, MH / 128), gblk, 0, stream>>>(
      ab16, wo_u, nullptr, out_und, HD, HIDc, 2, 0, 1, 0);
  gemm_bias_wmma<<<dim3(HIDc / 128, MH / 128), gblk, 0, stream>>>(
      ab16, wo_g, nullptr, out_gen, HD, HIDc, 2, 1, 1, 0);
}
